// GraphEncoder_33114197852466
// MI455X (gfx1250) — matrix operand order repaired and checked by reading, not compiled, not run on hardware
//
#include <hip/hip_runtime.h>

typedef __attribute__((ext_vector_type(16))) __bf16 v16bf;
typedef __attribute__((ext_vector_type(8)))  float  v8f;

#define HD 128

static inline int cdiv(long a, long b) { return (int)((a + b - 1) / b); }

// ---------------------------------------------------------------------------
// Transpose + convert a 128x128 f32 weight (row-major [K,N]) to bf16 [N,K].
// ---------------------------------------------------------------------------
__global__ void k_trans_bf16(const float* __restrict__ W, __bf16* __restrict__ Wt) {
    int k = blockIdx.x * 16 + threadIdx.x;
    int n = blockIdx.y * 16 + threadIdx.y;
    Wt[(size_t)n * HD + k] = (__bf16)W[(size_t)k * HD + n];
}

// ---------------------------------------------------------------------------
// GEMM: C[nrows,128] = A[nrows,128] (f32) x B[128,128], B given transposed as
// bf16 Bt[n][k]. One wave computes a 16-row x 128-col tile via 32 WMMAs.
// Optional bias + relu epilogue.
// A fragment layout (16x32 bf16): lane<16 -> row=lane, K = kb+{0..15};
// lane>=16 -> row=lane-16, K = kb+{16..31}.
// B fragment layout (32x16 bf16): lane holds col=lane%16, 16 contiguous K
// starting at kb + (lane>=16 ? 16 : 0).
// ---------------------------------------------------------------------------
__global__ void k_gemm_bf16(const float* __restrict__ A, const __bf16* __restrict__ Bt,
                            const float* __restrict__ bias, float* __restrict__ C,
                            int nrows, int do_relu) {
    int wave   = blockIdx.x * (blockDim.x >> 5) + (threadIdx.x >> 5);
    int ntiles = (nrows + 15) >> 4;
    if (wave >= ntiles) return;          // whole-wave uniform exit: EXEC all 1s for WMMA
    int lane = threadIdx.x & 31;
    int lr   = lane & 15;
    int hi   = lane >> 4;
    int row0 = wave << 4;

    int ar = row0 + lr; if (ar >= nrows) ar = nrows - 1;   // clamp (N%16==0 normally)
    const float* arow = A + (size_t)ar * HD;

    v8f acc[8] = {};

    #pragma unroll
    for (int kt = 0; kt < 4; ++kt) {
        int kb = kt << 5;
        const float4* ap0 = (const float4*)(arow + kb + hi * 16);
        const float4* ap1 = (const float4*)(arow + kb + hi * 16 + 8);
        float4 a0 = ap0[0], a1 = ap0[1], a2 = ap1[0], a3 = ap1[1];
        v16bf af;
        af[0]  = (__bf16)a0.x; af[1]  = (__bf16)a0.y; af[2]  = (__bf16)a0.z; af[3]  = (__bf16)a0.w;
        af[4]  = (__bf16)a1.x; af[5]  = (__bf16)a1.y; af[6]  = (__bf16)a1.z; af[7]  = (__bf16)a1.w;
        af[8]  = (__bf16)a2.x; af[9]  = (__bf16)a2.y; af[10] = (__bf16)a2.z; af[11] = (__bf16)a2.w;
        af[12] = (__bf16)a3.x; af[13] = (__bf16)a3.y; af[14] = (__bf16)a3.z; af[15] = (__bf16)a3.w;

        #pragma unroll
        for (int ct = 0; ct < 8; ++ct) {
            const v16bf* bp = (const v16bf*)(Bt + (size_t)(ct * 16 + lr) * HD + kb + hi * 16);
            v16bf bf = *bp;
            acc[ct] = __builtin_amdgcn_wmma_f32_16x16x32_bf16(
                false, af, false, bf, (short)0, acc[ct], false, false);
        }
    }

    // C/D layout: VGPR i, lane<16 -> row=i, col=lane; lane>=16 -> row=8+i, col=lane-16
    #pragma unroll
    for (int ct = 0; ct < 8; ++ct) {
        int c = ct * 16 + lr;
        float bv = bias ? bias[c] : 0.0f;
        #pragma unroll
        for (int i = 0; i < 8; ++i) {
            int r = row0 + hi * 8 + i;
            if (r < nrows) {
                float v = acc[ct][i] + bv;
                if (do_relu) v = v > 0.0f ? v : 0.0f;
                C[(size_t)r * HD + c] = v;
            }
        }
    }
}

// ---------------------------------------------------------------------------
// Degree / normalization
// ---------------------------------------------------------------------------
__global__ void k_fill_f32(float* p, float v, int n) {
    int i = blockIdx.x * blockDim.x + threadIdx.x;
    if (i < n) p[i] = v;
}
__global__ void k_deg_accum(const int* __restrict__ ei, int E, float* __restrict__ deg) {
    int i = blockIdx.x * blockDim.x + threadIdx.x;
    if (i < E) atomicAdd(&deg[ei[E + i]], 1.0f);
}
__global__ void k_rsqrt_inplace(float* p, int n) {
    int i = blockIdx.x * blockDim.x + threadIdx.x;
    if (i < n) p[i] = rsqrtf(p[i]);
}

// ---------------------------------------------------------------------------
// GCN aggregation: agg[d] += dinv[s]*dinv[d]*m[s]; self-loop folded into init.
// ---------------------------------------------------------------------------
__global__ void k_agg_init(const float* __restrict__ m, const float* __restrict__ dinv,
                           float* __restrict__ agg, int n) {
    size_t i = (size_t)blockIdx.x * blockDim.x + threadIdx.x;
    if (i < (size_t)n * HD) {
        int node = (int)(i >> 7);
        float w = dinv[node];
        agg[i] = w * w * m[i];
    }
}
__global__ void k_agg_edges(const float* __restrict__ m, const float* __restrict__ dinv,
                            const int* __restrict__ ei, int E, float* __restrict__ agg) {
    size_t t = (size_t)blockIdx.x * blockDim.x + threadIdx.x;
    int e = (int)(t >> 5);
    if (e >= E) return;
    int lane = threadIdx.x & 31;
    int s = ei[e], d = ei[E + e];
    float w = dinv[s] * dinv[d];
    float4 mv = ((const float4*)(m + (size_t)s * HD))[lane];
    float* dst = agg + (size_t)d * HD + lane * 4;
    atomicAdd(dst + 0, w * mv.x);
    atomicAdd(dst + 1, w * mv.y);
    atomicAdd(dst + 2, w * mv.z);
    atomicAdd(dst + 3, w * mv.w);
}
__global__ void k_bias_relu(float* __restrict__ h, const float* __restrict__ b, int n) {
    size_t i = (size_t)blockIdx.x * blockDim.x + threadIdx.x;
    if (i < (size_t)n * HD) {
        float v = h[i] + b[(int)(i & 127)];
        h[i] = v > 0.0f ? v : 0.0f;
    }
}

// ---------------------------------------------------------------------------
// GAT: per-node attention scalars, segment softmax over incoming edges.
// ---------------------------------------------------------------------------
__global__ void k_dots(const float* __restrict__ g, const float* __restrict__ a_src,
                       const float* __restrict__ a_dst, float* __restrict__ gs,
                       float* __restrict__ gd, int n) {
    int wave = (int)(((size_t)blockIdx.x * blockDim.x + threadIdx.x) >> 5);
    if (wave >= n) return;
    int lane = threadIdx.x & 31;
    float4 gv = ((const float4*)(g + (size_t)wave * HD))[lane];
    float4 as = ((const float4*)a_src)[lane];
    float4 ad = ((const float4*)a_dst)[lane];
    float ps = gv.x * as.x + gv.y * as.y + gv.z * as.z + gv.w * as.w;
    float pd = gv.x * ad.x + gv.y * ad.y + gv.z * ad.z + gv.w * ad.w;
    for (int off = 16; off; off >>= 1) {
        ps += __shfl_xor(ps, off);
        pd += __shfl_xor(pd, off);
    }
    if (lane == 0) { gs[wave] = ps; gd[wave] = pd; }
}

__device__ __forceinline__ int enc_f(float x) {
    int i = __float_as_int(x);
    return i >= 0 ? i : (i ^ 0x7FFFFFFF);
}
__device__ __forceinline__ float dec_f(int i) {
    return __int_as_float(i >= 0 ? i : (i ^ 0x7FFFFFFF));
}
__device__ __forceinline__ void edge_sd(const int* ei, int E, int idx, int& s, int& d) {
    if (idx < E) { s = ei[idx]; d = ei[E + idx]; }
    else         { s = d = idx - E; }          // self-loop
}

__global__ void k_gat_init(int* __restrict__ emax, float* __restrict__ denom, int n) {
    int i = blockIdx.x * blockDim.x + threadIdx.x;
    if (i < n) { emax[i] = (int)0x80000000; denom[i] = 0.0f; }
}
__global__ void k_gat_max(const int* __restrict__ ei, int E, int n,
                          const float* __restrict__ gs, const float* __restrict__ gd,
                          int* __restrict__ emax) {
    int i = blockIdx.x * blockDim.x + threadIdx.x;
    if (i >= E + n) return;
    int s, d; edge_sd(ei, E, i, s, d);
    float e = gs[s] + gd[d];
    e = e > 0.0f ? e : 0.2f * e;
    atomicMax(&emax[d], enc_f(e));
}
__global__ void k_gat_denom(const int* __restrict__ ei, int E, int n,
                            const float* __restrict__ gs, const float* __restrict__ gd,
                            const int* __restrict__ emax, float* __restrict__ denom) {
    int i = blockIdx.x * blockDim.x + threadIdx.x;
    if (i >= E + n) return;
    int s, d; edge_sd(ei, E, i, s, d);
    float e = gs[s] + gd[d];
    e = e > 0.0f ? e : 0.2f * e;
    atomicAdd(&denom[d], __expf(e - dec_f(emax[d])));
}
__global__ void k_zero_nh(float* p, int n) {
    size_t i = (size_t)blockIdx.x * blockDim.x + threadIdx.x;
    if (i < (size_t)n * HD) p[i] = 0.0f;
}
__global__ void k_gat_scatter(const float* __restrict__ g, const int* __restrict__ ei,
                              int E, int n, const float* __restrict__ gs,
                              const float* __restrict__ gd, const int* __restrict__ emax,
                              const float* __restrict__ denom, float* __restrict__ outacc) {
    size_t t = (size_t)blockIdx.x * blockDim.x + threadIdx.x;
    int idx = (int)(t >> 5);
    if (idx >= E + n) return;
    int lane = threadIdx.x & 31;
    int s, d; edge_sd(ei, E, idx, s, d);
    float e = gs[s] + gd[d];
    e = e > 0.0f ? e : 0.2f * e;
    float alpha = __expf(e - dec_f(emax[d])) / denom[d];
    float4 gv = ((const float4*)(g + (size_t)s * HD))[lane];
    float* dst = outacc + (size_t)d * HD + lane * 4;
    atomicAdd(dst + 0, alpha * gv.x);
    atomicAdd(dst + 1, alpha * gv.y);
    atomicAdd(dst + 2, alpha * gv.z);
    atomicAdd(dst + 3, alpha * gv.w);
}

// ---------------------------------------------------------------------------
// Final global mean over nodes -> out[1,128]
// ---------------------------------------------------------------------------
__global__ void k_zero_out(float* out) { out[threadIdx.x] = 0.0f; }
__global__ void k_colsum(const float* __restrict__ acc, float* __restrict__ out, int n) {
    int col = threadIdx.x & 127;
    int r0  = blockIdx.x * 2 + (threadIdx.x >> 7);
    float sum = 0.0f;
    for (int r = r0; r < n; r += gridDim.x * 2)
        sum += acc[(size_t)r * HD + col];
    atomicAdd(&out[col], sum);
}
__global__ void k_out_final(float* __restrict__ out, const float* __restrict__ bg, float invN) {
    int j = threadIdx.x;
    out[j] = out[j] * invN + bg[j];
}

// ---------------------------------------------------------------------------
extern "C" void kernel_launch(void* const* d_in, const int* in_sizes, int n_in,
                              void* d_out, int out_size, void* d_ws, size_t ws_size,
                              hipStream_t stream) {
    const float* x     = (const float*)d_in[0];
    const int*   ei    = (const int*)  d_in[1];
    const float* W_emb = (const float*)d_in[2];
    const float* b_emb = (const float*)d_in[3];
    const float* W1    = (const float*)d_in[4];
    const float* b1    = (const float*)d_in[5];
    const float* W2    = (const float*)d_in[6];
    const float* b2    = (const float*)d_in[7];
    const float* Wg    = (const float*)d_in[8];
    const float* a_src = (const float*)d_in[9];
    const float* a_dst = (const float*)d_in[10];
    const float* bg    = (const float*)d_in[11];
    float* out = (float*)d_out;

    const int N = in_sizes[0] / HD;
    const int E = in_sizes[1] / 2;

    // ---- workspace carve-up -------------------------------------------------
    char* ws = (char*)d_ws;
    size_t off = 0;
    auto take = [&](size_t bytes) { void* p = ws + off; off += (bytes + 255) & ~(size_t)255; return p; };
    const size_t NHb = (size_t)N * HD * sizeof(float);
    float* bufA  = (float*)take(NHb);
    float* bufB  = (float*)take(NHb);
    float* bufC  = (float*)take(NHb);
    float* dinv  = (float*)take((size_t)N * 4);   // deg, then rsqrt in place
    float* gs    = (float*)take((size_t)N * 4);
    float* gd    = (float*)take((size_t)N * 4);
    int*   emax  = (int*)  take((size_t)N * 4);
    float* denom = (float*)take((size_t)N * 4);
    __bf16* WtE  = (__bf16*)take((size_t)HD * HD * 2);
    __bf16* Wt1  = (__bf16*)take((size_t)HD * HD * 2);
    __bf16* Wt2  = (__bf16*)take((size_t)HD * HD * 2);
    __bf16* Wtg  = (__bf16*)take((size_t)HD * HD * 2);

    dim3 tb(16, 16), tg(8, 8);
    const int TPB = 256;
    const int gN    = cdiv(N, TPB);
    const int gNH   = cdiv((long)N * HD, TPB);
    const int gE    = cdiv(E, TPB);
    const int gEN   = cdiv((long)E + N, TPB);
    const int gEw   = cdiv((long)E * 32, TPB);
    const int gENw  = cdiv(((long)E + N) * 32, TPB);
    const int gGemm = cdiv(cdiv(N, 16), 8);

    // weights -> bf16 transposed
    k_trans_bf16<<<tg, tb, 0, stream>>>(W_emb, WtE);
    k_trans_bf16<<<tg, tb, 0, stream>>>(W1,    Wt1);
    k_trans_bf16<<<tg, tb, 0, stream>>>(W2,    Wt2);
    k_trans_bf16<<<tg, tb, 0, stream>>>(Wg,    Wtg);

    // degrees (self-loop => start at 1), dinv = rsqrt(deg)
    k_fill_f32<<<gN, TPB, 0, stream>>>(dinv, 1.0f, N);
    k_deg_accum<<<gE, TPB, 0, stream>>>(ei, E, dinv);
    k_rsqrt_inplace<<<gN, TPB, 0, stream>>>(dinv, N);

    // h0 = relu(x @ W_emb + b_emb)            -> bufA
    k_gemm_bf16<<<gGemm, TPB, 0, stream>>>(x, WtE, b_emb, bufA, N, 1);

    // GCN layer 1: m = h0 @ W1 -> bufB ; agg -> bufC ; h1 = relu(agg+b1) in bufC
    k_gemm_bf16<<<gGemm, TPB, 0, stream>>>(bufA, Wt1, nullptr, bufB, N, 0);
    k_agg_init<<<gNH, TPB, 0, stream>>>(bufB, dinv, bufC, N);
    k_agg_edges<<<gEw, TPB, 0, stream>>>(bufB, dinv, ei, E, bufC);
    k_bias_relu<<<gNH, TPB, 0, stream>>>(bufC, b1, N);

    // GCN layer 2: m = h1 @ W2 -> bufB ; agg -> bufA ; h2 = relu(agg+b2) in bufA
    k_gemm_bf16<<<gGemm, TPB, 0, stream>>>(bufC, Wt2, nullptr, bufB, N, 0);
    k_agg_init<<<gNH, TPB, 0, stream>>>(bufB, dinv, bufA, N);
    k_agg_edges<<<gEw, TPB, 0, stream>>>(bufB, dinv, ei, E, bufA);
    k_bias_relu<<<gNH, TPB, 0, stream>>>(bufA, b2, N);

    // GAT: g = h2 @ Wg -> bufB
    k_gemm_bf16<<<gGemm, TPB, 0, stream>>>(bufA, Wtg, nullptr, bufB, N, 0);
    k_dots<<<cdiv((long)N * 32, TPB), TPB, 0, stream>>>(bufB, a_src, a_dst, gs, gd, N);
    k_gat_init<<<gN, TPB, 0, stream>>>(emax, denom, N);
    k_gat_max<<<gEN, TPB, 0, stream>>>(ei, E, N, gs, gd, emax);
    k_gat_denom<<<gEN, TPB, 0, stream>>>(ei, E, N, gs, gd, emax, denom);
    k_zero_nh<<<gNH, TPB, 0, stream>>>(bufC, N);
    k_gat_scatter<<<gENw, TPB, 0, stream>>>(bufB, ei, E, N, gs, gd, emax, denom, bufC);

    // global mean + bg
    k_zero_out<<<1, HD, 0, stream>>>(out);
    k_colsum<<<256, TPB, 0, stream>>>(bufC, out, N);
    k_out_final<<<1, HD, 0, stream>>>(out, bg, 1.0f / (float)N);

    (void)n_in; (void)out_size; (void)ws_size;
}
